// DyGATFR_88742614270131
// MI455X (gfx1250) — matrix operand order repaired and checked by reading, not compiled, not run on hardware
//
#include <hip/hip_runtime.h>
#include <math.h>

// ---------------------------------------------------------------------------
// DyGAT forward for MI455X (gfx1250).
// Dense matmuls: v_wmma_f32_16x16x32_bf16 with pre-packed bf16 operands
// (activations converted once per GEMM, weights transposed+converted once
// per launch) so the WMMA inner loop is pure b128 loads + wmma.
// Edge softmax: 3 streaming atomic passes. Node-wise ops: wave32 kernels.
// ---------------------------------------------------------------------------

typedef __attribute__((ext_vector_type(16))) __bf16 v16bf;
typedef __attribute__((ext_vector_type(8)))  __bf16 v8bf;
typedef __attribute__((ext_vector_type(8)))  float  v8f;

#define HEADS 4

static __device__ __forceinline__ __bf16 f2bf(float f) {
  unsigned u = __float_as_uint(f);
  u += 0x7FFFu + ((u >> 16) & 1u);          // round-to-nearest-even
  unsigned short s = (unsigned short)(u >> 16);
  __bf16 r;
  __builtin_memcpy(&r, &s, sizeof(r));
  return r;
}

static __device__ __forceinline__ float gelu_f(float x) {
  return 0.5f * x * (1.f + erff(x * 0.70710678118654752f));
}

static __device__ __forceinline__ float sigmoid_f(float x) {
  return 1.f / (1.f + expf(-x));
}

static __device__ __forceinline__ void atomicMaxF(float* addr, float v) {
  if (v >= 0.f) atomicMax((int*)addr, __float_as_int(v));
  else          atomicMin((unsigned int*)addr, __float_as_uint(v));
}

// ---------------------------------------------------------------------------
// fp32 -> bf16 bulk convert (row-major, same layout)
__global__ void k_f2bf(const float* __restrict__ in, __bf16* __restrict__ out,
                       long long n) {
  long long i = (long long)blockIdx.x * blockDim.x + threadIdx.x;
  if (i < n) out[i] = f2bf(in[i]);
}

// transpose + convert weights: W[K,NC] f32 -> Wt[NC,K] bf16
__global__ void k_pack_w(const float* __restrict__ W, __bf16* __restrict__ Wt,
                         int K, int NC) {
  int t = blockIdx.x * blockDim.x + threadIdx.x;
  if (t >= K * NC) return;
  int c = t / K, k = t - c * K;
  Wt[(size_t)c * K + k] = f2bf(W[(size_t)k * NC + c]);
}

// ---------------------------------------------------------------------------
// WMMA GEMM: C[M,NC] = A[M,K] @ W[K,NC] (+bias), A bf16 row-major,
// Wt bf16 column-major (NC x K). One wave per 16x16 C tile, 4 waves/block.
// M%16==0, K%32==0, NC%16==0.
//
// Fragment maps (ISA 7.12.2, ext-vector elem i -> VGPR i/2 lo/hi):
//   A: lane half -> kbase{0,16}; a[i]=K kbase+i, i<16 (contig, same as B)
//   B: lanes 0-15 K 0..15, lanes 16-31 K 16..31; b[i]=K half*16+i (contig)
//   C: VGPR v -> row half*8+v, col=l16
// ---------------------------------------------------------------------------
__global__ void k_gemm_wmma(const __bf16* __restrict__ A, const __bf16* __restrict__ Wt,
                            const float* __restrict__ bias, float* __restrict__ C,
                            int M, int K, int NC) {
  int rt = blockIdx.x * blockDim.y + threadIdx.y;     // row tile
  if (rt * 16 >= M) return;
  int lane = threadIdx.x;                             // 0..31 (wave32)
  int half = lane >> 4;
  int l16  = lane & 15;
  int row0 = rt * 16;
  int col  = blockIdx.y * 16 + l16;

  const __bf16* arow = A + (size_t)(row0 + l16) * K + half * 16;
  const __bf16* wcol = Wt + (size_t)col * K + half * 16;

  v8f acc = {};
  for (int kk = 0; kk < K; kk += 32) {
    v8bf a0 = *(const v8bf*)(arow + kk);
    v8bf a1 = *(const v8bf*)(arow + kk + 8);
    v16bf a = __builtin_shufflevector(a0, a1, 0, 1, 2, 3, 4, 5, 6, 7,
                                      8, 9, 10, 11, 12, 13, 14, 15);
    v16bf b = *(const v16bf*)(wcol + kk);
    acc = __builtin_amdgcn_wmma_f32_16x16x32_bf16(
        false, a, false, b, (short)0, acc, false, false);
  }
  float bv = bias ? bias[col] : 0.f;
#pragma unroll
  for (int v = 0; v < 8; ++v)
    C[(size_t)(row0 + half * 8 + v) * NC + col] = acc[v] + bv;
}

// ---------------------------------------------------------------------------
__global__ void k_fill(float* __restrict__ p, float v, long long n) {
  long long i = (long long)blockIdx.x * blockDim.x + threadIdx.x;
  if (i < n) p[i] = v;
}

// p_init / focal / focal_weights from pe hidden (t, 64 wide, pre-relu)
__global__ void k_pe(const float* __restrict__ t, const float* __restrict__ w2,
                     const float* __restrict__ b2, float* __restrict__ p_init,
                     float* __restrict__ focal, float* __restrict__ fw, int N) {
  int n = blockIdx.x * blockDim.x + threadIdx.x;
  if (n >= N) return;
  const float* tr = t + (size_t)n * 64;
  float acc = b2[0];
#pragma unroll 8
  for (int j = 0; j < 64; ++j) acc += fmaxf(tr[j], 0.f) * w2[j];
  float p = sigmoid_f(acc);
  p_init[n] = p;
  float omp = 1.f - p;
  focal[n] = 0.25f * omp * omp;              // F_ALPHA * (1-p)^F_GAMMA
  fw[n] = 1.f - 2.f * fabsf(p - 0.5f);
}

// LayerNorm + exact GELU (+optional residual). One wave per node.
__global__ void k_ln_gelu(const float* __restrict__ Y, const float* __restrict__ g,
                          const float* __restrict__ b, const float* __restrict__ res,
                          float* __restrict__ O, int N, int D) {
  int node = blockIdx.x * blockDim.y + threadIdx.y;
  if (node >= N) return;
  int lane = threadIdx.x;
  int nE = D >> 5;                            // 4 for D=128, 2 for D=64
  const float* y = Y + (size_t)node * D;
  float vals[4];
  float s = 0.f, q = 0.f;
  for (int j = 0; j < nE; ++j) {
    float v = y[j * 32 + lane];
    vals[j] = v; s += v; q += v * v;
  }
  for (int m = 16; m > 0; m >>= 1) {
    s += __shfl_xor(s, m, 32);
    q += __shfl_xor(q, m, 32);
  }
  float mean = s / (float)D;
  float var  = q / (float)D - mean * mean;
  float rs   = rsqrtf(var + 1e-5f);
  float* o = O + (size_t)node * D;
  const float* r = res ? res + (size_t)node * D : nullptr;
  for (int j = 0; j < nE; ++j) {
    int c = j * 32 + lane;
    float v = (vals[j] - mean) * rs * g[c] + b[c];
    v = gelu_f(v);
    if (r) v += r[c];
    o[c] = v;
  }
}

// attention logits per (node, head)
__global__ void k_att(const float* __restrict__ xh, const float* __restrict__ wsrc,
                      const float* __restrict__ wdst, float* __restrict__ asrc,
                      float* __restrict__ adst, int N, int hd) {
  int t = blockIdx.x * blockDim.x + threadIdx.x;
  if (t >= N * HEADS) return;
  int n = t >> 2, h = t & 3;
  int D = HEADS * hd;
  const float* xr = xh + (size_t)n * D + h * hd;
  float ss = 0.f, sd = 0.f;
  for (int j = 0; j < hd; ++j) {
    float v = xr[j];
    ss += v * wsrc[h * hd + j];
    sd += v * wdst[h * hd + j];
  }
  asrc[t] = ss; adst[t] = sd;
}

// pass 1: e = leaky_relu(a_src[s]+a_dst[d]) * (1+focal[d]); segment max
// one thread per edge, all 4 heads (b128-merged loads)
__global__ void k_edge1(const int* __restrict__ src, const int* __restrict__ dst,
                        const float* __restrict__ asrc, const float* __restrict__ adst,
                        const float* __restrict__ focal, float* __restrict__ ebuf,
                        float* __restrict__ emax, int E) {
  int e = blockIdx.x * blockDim.x + threadIdx.x;
  if (e >= E) return;
  int s = src[e], d = dst[e];
  float fd = 1.f + focal[d];
  const float* as_ = asrc + (size_t)s * HEADS;
  const float* ad_ = adst + (size_t)d * HEADS;
#pragma unroll
  for (int h = 0; h < HEADS; ++h) {
    float ev = as_[h] + ad_[h];
    ev = (ev > 0.f) ? ev : 0.2f * ev;
    ev *= fd;
    ebuf[(size_t)e * HEADS + h] = ev;
    atomicMaxF(&emax[(size_t)d * HEADS + h], ev);
  }
}

// pass 2: ex = exp(e - emax[d]); segment sum
__global__ void k_edge2(const int* __restrict__ dst, float* __restrict__ ebuf,
                        const float* __restrict__ emax, float* __restrict__ denom,
                        int E) {
  int e = blockIdx.x * blockDim.x + threadIdx.x;
  if (e >= E) return;
  int d = dst[e];
#pragma unroll
  for (int h = 0; h < HEADS; ++h) {
    float ex = expf(ebuf[(size_t)e * HEADS + h] - emax[(size_t)d * HEADS + h]);
    ebuf[(size_t)e * HEADS + h] = ex;
    atomicAdd(&denom[(size_t)d * HEADS + h], ex);
  }
}

// pass 3: out[d] += alpha * xh[s]   (one thread per (edge, channel))
__global__ void k_edge3(const int* __restrict__ src, const int* __restrict__ dst,
                        const float* __restrict__ ebuf, const float* __restrict__ denom,
                        const float* __restrict__ xh, float* __restrict__ outb,
                        int E, int D, int hd) {
  long long t = (long long)blockIdx.x * blockDim.x + threadIdx.x;
  if (t >= (long long)E * D) return;
  int e = (int)(t / D);
  int c = (int)(t - (long long)e * D);
  int h = c / hd;
  int s = src[e], d = dst[e];
  float alpha = ebuf[e * HEADS + h] / (denom[d * HEADS + h] + 1e-16f);
  atomicAdd(&outb[(size_t)d * D + c], alpha * xh[(size_t)s * D + c]);
}

// final head: prototype attention + focal gate + gate MLP + classifier
__global__ void k_head(float* __restrict__ h, const float* __restrict__ q,
                       const float* __restrict__ p_init, const float* __restrict__ fw,
                       const float* __restrict__ pk, const float* __restrict__ pv,
                       const float* __restrict__ gw1, const float* __restrict__ gb1,
                       const float* __restrict__ gw2, const float* __restrict__ gb2,
                       const float* __restrict__ cw1, const float* __restrict__ cb1,
                       const float* __restrict__ cw2, const float* __restrict__ cb2,
                       float* __restrict__ out, int N) {
  int n = blockIdx.x * blockDim.x + threadIdx.x;
  if (n >= N) return;
  const float* qr = q + (size_t)n * 64;

  float sc[8], smax = -3.4e38f;
#pragma unroll
  for (int p = 0; p < 8; ++p) {
    float s = 0.f;
    for (int j = 0; j < 64; ++j) s += qr[j] * pk[p * 64 + j];
    sc[p] = s * 0.125f;                      // / sqrt(64)
    smax = fmaxf(smax, sc[p]);
  }
  float ssum = 0.f;
#pragma unroll
  for (int p = 0; p < 8; ++p) { sc[p] = expf(sc[p] - smax); ssum += sc[p]; }
  float inv = 1.f / ssum;
#pragma unroll
  for (int p = 0; p < 8; ++p) sc[p] *= inv;

  float f = fw[n];
  float hr[64];
  const float* hrow = h + (size_t)n * 64;
  for (int j = 0; j < 64; ++j) {
    float ctx = 0.f;
#pragma unroll
    for (int p = 0; p < 8; ++p) ctx += sc[p] * pv[p * 64 + j];
    hr[j] = hrow[j] + f * ctx;
  }

  // gate MLP: gin = [hr, p_init]  (65 -> 16 -> 64)
  float pin = p_init[n];
  float hid[16];
#pragma unroll
  for (int k = 0; k < 16; ++k) {
    float acc = gb1[k];
    for (int j = 0; j < 64; ++j) acc += hr[j] * gw1[j * 16 + k];
    acc += pin * gw1[64 * 16 + k];
    hid[k] = fmaxf(acc, 0.f);
  }
  for (int j = 0; j < 64; ++j) {
    float acc = gb2[j];
#pragma unroll
    for (int k = 0; k < 16; ++k) acc += hid[k] * gw2[k * 64 + j];
    hr[j] *= sigmoid_f(acc);
  }

  // classifier: 64 -> 32 (gelu) -> 1
  float logit = cb2[0];
  for (int k = 0; k < 32; ++k) {
    float acc = cb1[k];
    for (int j = 0; j < 64; ++j) acc += hr[j] * cw1[j * 32 + k];
    logit += gelu_f(acc) * cw2[k];
  }
  out[n] = logit;
}

// ---------------------------------------------------------------------------
static inline void gemm(const __bf16* A, const __bf16* Wt, const float* bias,
                        float* C, int M, int K, int NC, hipStream_t s) {
  dim3 blk(32, 4);
  dim3 grd((M / 16 + 3) / 4, NC / 16);
  k_gemm_wmma<<<grd, blk, 0, s>>>(A, Wt, bias, C, M, K, NC);
}

static inline void fill(float* p, float v, long long n, hipStream_t s) {
  k_fill<<<(unsigned)((n + 255) / 256), 256, 0, s>>>(p, v, n);
}

static inline void cvt(const float* in, __bf16* out, long long n, hipStream_t s) {
  k_f2bf<<<(unsigned)((n + 255) / 256), 256, 0, s>>>(in, out, n);
}

static inline void packw(const float* W, __bf16* Wt, int K, int NC, hipStream_t s) {
  k_pack_w<<<(K * NC + 255) / 256, 256, 0, s>>>(W, Wt, K, NC);
}

extern "C" void kernel_launch(void* const* d_in, const int* in_sizes, int n_in,
                              void* d_out, int out_size, void* d_ws, size_t ws_size,
                              hipStream_t stream) {
  const float* x  = (const float*)d_in[0];          // (N,256)
  const int*   ei = (const int*)d_in[1];            // (2,E)
  const int N = in_sizes[0] / 256;
  const int E = in_sizes[1] / 2;
  const int* src = ei;
  const int* dst = ei + E;

  // params, setup_inputs() dict insertion order, convs depth-first
  const float* pe_w1 = (const float*)d_in[2];   // 256x64
  const float* pe_b1 = (const float*)d_in[3];
  const float* pe_w2 = (const float*)d_in[4];   // 64
  const float* pe_b2 = (const float*)d_in[5];
  const float* in_w  = (const float*)d_in[6];   // 256x128
  const float* in_b  = (const float*)d_in[7];
  const float* in_g  = (const float*)d_in[8];
  const float* in_be = (const float*)d_in[9];
  const float* lin[3]  = {(const float*)d_in[10], (const float*)d_in[15], (const float*)d_in[20]};
  const float* asw[3]  = {(const float*)d_in[11], (const float*)d_in[16], (const float*)d_in[21]};
  const float* adw[3]  = {(const float*)d_in[12], (const float*)d_in[17], (const float*)d_in[22]};
  const float* ng[3]   = {(const float*)d_in[13], (const float*)d_in[18], (const float*)d_in[23]};
  const float* nb[3]   = {(const float*)d_in[14], (const float*)d_in[19], (const float*)d_in[24]};
  const float* res_w = (const float*)d_in[25];  // 128x64
  const float* res_b = (const float*)d_in[26];
  const float* pk    = (const float*)d_in[27];  // 8x64
  const float* pv    = (const float*)d_in[28];  // 8x64
  const float* q_w   = (const float*)d_in[29];  // 64x64
  const float* q_b   = (const float*)d_in[30];
  const float* gw1   = (const float*)d_in[31];  // 65x16
  const float* gb1   = (const float*)d_in[32];
  const float* gw2   = (const float*)d_in[33];  // 16x64
  const float* gb2   = (const float*)d_in[34];
  const float* cw1   = (const float*)d_in[35];  // 64x32
  const float* cb1   = (const float*)d_in[36];
  const float* cw2   = (const float*)d_in[37];  // 32
  const float* cb2   = (const float*)d_in[38];

  // workspace carve (bytes)
  char* wsb = (char*)d_ws;
  size_t o = 0;
  auto alloc = [&](size_t bytes) { char* p = wsb + o; o += (bytes + 255) & ~(size_t)255; return p; };
  float*  h      = (float*)alloc((size_t)N * 128 * 4);
  float*  xh     = (float*)alloc((size_t)N * 128 * 4);
  float*  outb   = (float*)alloc((size_t)N * 128 * 4);
  float*  ebuf   = (float*)alloc((size_t)E * HEADS * 4);
  float*  t64    = (float*)alloc((size_t)N * 64 * 4);   // pe-hidden / res / q
  float*  p_init = (float*)alloc((size_t)N * 4);
  float*  focal  = (float*)alloc((size_t)N * 4);
  float*  fw     = (float*)alloc((size_t)N * 4);
  float*  a_src  = (float*)alloc((size_t)N * HEADS * 4);
  float*  a_dst  = (float*)alloc((size_t)N * HEADS * 4);
  float*  emax   = (float*)alloc((size_t)N * HEADS * 4);
  float*  denom  = (float*)alloc((size_t)N * HEADS * 4);
  __bf16* xbf    = (__bf16*)alloc((size_t)N * 256 * 2);
  __bf16* hbf    = (__bf16*)alloc((size_t)N * 128 * 2);
  __bf16* pe_w1t = (__bf16*)alloc(256 * 64 * 2);
  __bf16* in_wt  = (__bf16*)alloc(256 * 128 * 2);
  __bf16* lint[3];
  lint[0] = (__bf16*)alloc(128 * 128 * 2);
  lint[1] = (__bf16*)alloc(128 * 128 * 2);
  lint[2] = (__bf16*)alloc(128 * 64 * 2);
  __bf16* res_wt = (__bf16*)alloc(128 * 64 * 2);
  __bf16* q_wt   = (__bf16*)alloc(64 * 64 * 2);
  (void)ws_size; (void)n_in; (void)out_size;

  const int TB = 256;

  // ---- pack weights (bf16, transposed) and input activations
  packw(pe_w1, pe_w1t, 256, 64, stream);
  packw(in_w, in_wt, 256, 128, stream);
  packw(lin[0], lint[0], 128, 128, stream);
  packw(lin[1], lint[1], 128, 128, stream);
  packw(lin[2], lint[2], 128, 64, stream);
  packw(res_w, res_wt, 128, 64, stream);
  packw(q_w, q_wt, 64, 64, stream);
  cvt(x, xbf, (long long)N * 256, stream);

  // ---- PE branch: p_init, focal, focal_weights
  gemm(xbf, pe_w1t, pe_b1, t64, N, 256, 64, stream);
  k_pe<<<(N + TB - 1) / TB, TB, 0, stream>>>(t64, pe_w2, pe_b2, p_init, focal, fw, N);

  // ---- input proj: h = gelu(ln(x @ in_w + in_b))
  gemm(xbf, in_wt, in_b, outb, N, 256, 128, stream);
  {
    dim3 blk(32, 4), grd((N + 3) / 4);
    k_ln_gelu<<<grd, blk, 0, stream>>>(outb, in_g, in_be, nullptr, h, N, 128);
  }

  // ---- 3 DyGAT conv layers
  const int hds[3] = {32, 32, 16};
  for (int i = 0; i < 3; ++i) {
    int hd = hds[i], D = HEADS * hd;
    cvt(h, hbf, (long long)N * 128, stream);
    gemm(hbf, lint[i], nullptr, xh, N, 128, D, stream);
    k_att<<<(N * HEADS + TB - 1) / TB, TB, 0, stream>>>(xh, asw[i], adw[i], a_src, a_dst, N, hd);
    fill(emax, -3.4e38f, (long long)N * HEADS, stream);
    fill(denom, 0.f, (long long)N * HEADS, stream);
    fill(outb, 0.f, (long long)N * D, stream);
    k_edge1<<<(E + TB - 1) / TB, TB, 0, stream>>>(src, dst, a_src, a_dst, focal, ebuf, emax, E);
    k_edge2<<<(E + TB - 1) / TB, TB, 0, stream>>>(dst, ebuf, emax, denom, E);
    {
      long long tot = (long long)E * D;
      k_edge3<<<(unsigned)((tot + TB - 1) / TB), TB, 0, stream>>>(src, dst, ebuf, denom, xh, outb, E, D, hd);
    }
    const float* res = nullptr;
    if (i < 2) {
      res = h;                                   // same width (128): identity residual
    } else {
      gemm(hbf, res_wt, res_b, t64, N, 128, 64, stream);  // projected residual
      res = t64;
    }
    dim3 blk(32, 4), grd((N + 3) / 4);
    k_ln_gelu<<<grd, blk, 0, stream>>>(outb, ng[i], nb[i], res, h, N, D);
  }

  // ---- head: q = h @ q_w + q_b; prototype attention + gate + classifier
  cvt(h, hbf, (long long)N * 64, stream);
  gemm(hbf, q_wt, q_b, t64, N, 64, 64, stream);
  k_head<<<(N + TB - 1) / TB, TB, 0, stream>>>(h, t64, p_init, fw, pk, pv,
                                               gw1, gb1, gw2, gb2,
                                               cw1, cb1, cw2, cb2,
                                               (float*)d_out, N);
}
